// CosineVectorQuantizer_62577673503565
// MI455X (gfx1250) — matrix operand order repaired and checked by reading, not compiled, not run on hardware
//
#include <hip/hip_runtime.h>
#include <hip/hip_bf16.h>

typedef __attribute__((ext_vector_type(16))) __bf16 v16bf;
typedef __attribute__((ext_vector_type(8)))  __bf16 v8bf;
typedef __attribute__((ext_vector_type(8)))  float  v8f;
typedef __attribute__((ext_vector_type(4)))  int    v4i;

#define B_ROWS  32768
#define DIM     256
#define N_CODES 8192
#define BETA_C  0.25f

#define LDS_ROW     264     // 256 bf16 + 8 pad -> 528B row stride, conflict-free b128 reads
#define SUPER_COLS  32      // codes staged per LDS buffer
#define NSPLIT      4       // N-dimension grid split
#define CODES_PER_SPLIT (N_CODES / NSPLIT)        // 2048
#define NSUP        (CODES_PER_SPLIT / SUPER_COLS) // 64

#if defined(__HIP_DEVICE_COMPILE__) && __has_builtin(__builtin_amdgcn_global_load_async_to_lds_b128)
#define HAVE_ASYNC_LDS 1
#else
#define HAVE_ASYNC_LDS 0
#endif

__device__ __forceinline__ void wait_async_done() {
#if HAVE_ASYNC_LDS
#if __has_builtin(__builtin_amdgcn_s_wait_asynccnt)
    __builtin_amdgcn_s_wait_asynccnt(0);
#else
    asm volatile("s_wait_asynccnt 0" ::: "memory");
#endif
#endif
}

// Stage SUPER_COLS codebook rows (bf16, row-major) into a padded LDS buffer.
// 256 threads x 4 chunks of 16B each = 32 rows * 512B.
__device__ __forceinline__ void stage_super(const __bf16* __restrict__ src_rows,
                                            __bf16* lbuf, int tid) {
    #pragma unroll
    for (int i = 0; i < 4; ++i) {
        const int c   = tid + i * 256;      // 0..1023
        const int row = c >> 5;             // 0..31 (32 x 16B chunks per row)
        const int col = (c & 31) * 8;       // element offset
        const __bf16* g = src_rows + row * DIM + col;
        __bf16* l = lbuf + row * LDS_ROW + col;
#if HAVE_ASYNC_LDS
        v4i* gg = (v4i*)const_cast<__bf16*>(g);   // generic, int4 pointee
        v4i* ll = (v4i*)l;
        __builtin_amdgcn_global_load_async_to_lds_b128(
            (__attribute__((address_space(1))) v4i*)gg,
            (__attribute__((address_space(3))) v4i*)ll, 0, 0);
#else
        *(v8bf*)l = *(const v8bf*)g;
#endif
    }
}

// ---------------------------------------------------------------------------
// Kernel 1: l2-normalize codebook rows -> bf16 row-major (N x D) in ws.
// ---------------------------------------------------------------------------
__global__ __launch_bounds__(256)
void k_norm_codebook(const float* __restrict__ emb, unsigned short* __restrict__ cbn_raw) {
    __bf16* cbn = reinterpret_cast<__bf16*>(cbn_raw);
    const int wave = threadIdx.x >> 5;
    const int lane = threadIdx.x & 31;
    const int row  = blockIdx.x * 8 + wave;
    const float* src = emb + (size_t)row * DIM + lane * 8;
    float4 a = *(const float4*)(src);
    float4 b = *(const float4*)(src + 4);
    float ss = a.x*a.x + a.y*a.y + a.z*a.z + a.w*a.w
             + b.x*b.x + b.y*b.y + b.z*b.z + b.w*b.w;
    #pragma unroll
    for (int off = 16; off; off >>= 1) ss += __shfl_xor(ss, off);
    const float scale = 1.0f / fmaxf(sqrtf(ss), 1e-12f);
    __bf16* dst = cbn + (size_t)row * DIM + lane * 8;
    v8bf o;
    o[0] = (__bf16)(a.x * scale); o[1] = (__bf16)(a.y * scale);
    o[2] = (__bf16)(a.z * scale); o[3] = (__bf16)(a.w * scale);
    o[4] = (__bf16)(b.x * scale); o[5] = (__bf16)(b.y * scale);
    o[6] = (__bf16)(b.z * scale); o[7] = (__bf16)(b.w * scale);
    *(v8bf*)dst = o;
}

// ---------------------------------------------------------------------------
// Kernel 2: fused x-normalize + WMMA sim GEMM + partial argmax.
// Wave owns M=32 (two 16-row A blocks, K=256 resident in 16 v16bf fragments).
// B super-tiles staged once per block into double-buffered LDS (async copy),
// each B fragment feeds two WMMAs. Grid: (B/256 row blocks) x (NSPLIT).
// ---------------------------------------------------------------------------
__global__ __launch_bounds__(256)
void k_sim_argmax(const float* __restrict__ x,
                  const unsigned short* __restrict__ cbn_raw,
                  float* __restrict__ partv,
                  int* __restrict__ partn) {
    __shared__ __bf16 lds_b[2][SUPER_COLS * LDS_ROW];
    const __bf16* cbn = reinterpret_cast<const __bf16*>(cbn_raw);
    const int tid  = threadIdx.x;
    const int wave = tid >> 5;
    const int lane = tid & 31;
    const int m       = lane & 15;
    const int hi_half = lane >> 4;
    const int rowBase  = blockIdx.x * 256 + wave * 32;
    const int codeBase = blockIdx.y * CODES_PER_SPLIT;
    const int khalfA = hi_half * 8;

    // Kick off first B super-tile while we build the A fragments.
    stage_super(cbn + (size_t)codeBase * DIM, lds_b[0], tid);

    v16bf afrag[2][8];
    #pragma unroll
    for (int a = 0; a < 2; ++a) {
        const float* xrow = x + (size_t)(rowBase + a * 16 + m) * DIM;
        float ss = 0.0f;
        #pragma unroll
        for (int kc = 0; kc < 8; ++kc) {
            const int k0 = kc * 32 + khalfA;
            float4 p0 = *(const float4*)(xrow + k0);
            float4 p1 = *(const float4*)(xrow + k0 + 4);
            float4 p2 = *(const float4*)(xrow + k0 + 16);
            float4 p3 = *(const float4*)(xrow + k0 + 20);
            ss += p0.x*p0.x + p0.y*p0.y + p0.z*p0.z + p0.w*p0.w;
            ss += p1.x*p1.x + p1.y*p1.y + p1.z*p1.z + p1.w*p1.w;
            ss += p2.x*p2.x + p2.y*p2.y + p2.z*p2.z + p2.w*p2.w;
            ss += p3.x*p3.x + p3.y*p3.y + p3.z*p3.z + p3.w*p3.w;
        }
        ss += __shfl_xor(ss, 16);
        const float scale = 1.0f / fmaxf(sqrtf(ss), 1e-12f);
        #pragma unroll
        for (int kc = 0; kc < 8; ++kc) {
            const int k0 = kc * 32 + khalfA;
            const float* p = xrow + k0;
            #pragma unroll
            for (int i = 0; i < 8; ++i) afrag[a][kc][i] = (__bf16)(p[i] * scale);
            p = xrow + k0 + 16;
            #pragma unroll
            for (int i = 0; i < 8; ++i) afrag[a][kc][8 + i] = (__bf16)(p[i] * scale);
        }
    }

    float bestv[2][8];
    int   bestn[2][8];
    #pragma unroll
    for (int a = 0; a < 2; ++a)
        #pragma unroll
        for (int r = 0; r < 8; ++r) { bestv[a][r] = -2.0f; bestn[a][r] = 0; }

    const int khalfB = hi_half * 8;
    wait_async_done();
    __syncthreads();

    for (int s = 0; s < NSUP; ++s) {
        const __bf16* buf = lds_b[s & 1];
        if (s + 1 < NSUP)
            stage_super(cbn + (size_t)(codeBase + (s + 1) * SUPER_COLS) * DIM,
                        lds_b[(s + 1) & 1], tid);
        #pragma unroll
        for (int sub = 0; sub < 2; ++sub) {
            const int ncol = codeBase + s * SUPER_COLS + sub * 16 + m;
            const __bf16* brow = buf + (sub * 16 + m) * LDS_ROW + khalfB;
            v8f c0 = {};
            v8f c1 = {};
            #pragma unroll
            for (int kc = 0; kc < 8; ++kc) {
                const v8bf lo = *(const v8bf*)(brow + kc * 32);
                const v8bf hi = *(const v8bf*)(brow + kc * 32 + 16);
                v16bf bfrag;
                #pragma unroll
                for (int i = 0; i < 8; ++i) { bfrag[i] = lo[i]; bfrag[8 + i] = hi[i]; }
                c0 = __builtin_amdgcn_wmma_f32_16x16x32_bf16(
                         false, afrag[0][kc], false, bfrag, (short)0, c0, false, false);
                c1 = __builtin_amdgcn_wmma_f32_16x16x32_bf16(
                         false, afrag[1][kc], false, bfrag, (short)0, c1, false, false);
            }
            #pragma unroll
            for (int r = 0; r < 8; ++r) {
                if (c0[r] > bestv[0][r]) { bestv[0][r] = c0[r]; bestn[0][r] = ncol; }
                if (c1[r] > bestv[1][r]) { bestv[1][r] = c1[r]; bestn[1][r] = ncol; }
            }
        }
        wait_async_done();
        __syncthreads();
    }

    // Reduce (max, lower-index-on-tie) across the 16 lanes sharing each row.
    #pragma unroll
    for (int a = 0; a < 2; ++a) {
        #pragma unroll
        for (int r = 0; r < 8; ++r) {
            #pragma unroll
            for (int off = 8; off; off >>= 1) {
                float ov = __shfl_xor(bestv[a][r], off);
                int   oi = __shfl_xor(bestn[a][r], off);
                if (ov > bestv[a][r] || (ov == bestv[a][r] && oi < bestn[a][r])) {
                    bestv[a][r] = ov; bestn[a][r] = oi;
                }
            }
        }
    }
    if (m == 0) {
        float* pv = partv + (size_t)blockIdx.y * B_ROWS;
        int*   pn = partn + (size_t)blockIdx.y * B_ROWS;
        #pragma unroll
        for (int a = 0; a < 2; ++a) {
            const int rb = rowBase + a * 16 + hi_half * 8;
            #pragma unroll
            for (int r = 0; r < 8; ++r) {
                pv[rb + r] = bestv[a][r];
                pn[rb + r] = bestn[a][r];
            }
        }
    }
}

// ---------------------------------------------------------------------------
// Kernel 3: merge the NSPLIT partial argmax candidates per row.
// ---------------------------------------------------------------------------
__global__ __launch_bounds__(256)
void k_merge(const float* __restrict__ partv, const int* __restrict__ partn,
             int* __restrict__ idx_i, float* __restrict__ idx_f) {
    const int row = blockIdx.x * 256 + threadIdx.x;
    float bv = partv[row];
    int   bn = partn[row];
    #pragma unroll
    for (int sp = 1; sp < NSPLIT; ++sp) {
        const float v = partv[(size_t)sp * B_ROWS + row];
        const int   n = partn[(size_t)sp * B_ROWS + row];
        if (v > bv || (v == bv && n < bn)) { bv = v; bn = n; }
    }
    idx_i[row] = bn;
    idx_f[row] = (float)bn;
}

// ---------------------------------------------------------------------------
// Kernel 4: gather selected code row; scalar projection, x_q, scalar out,
// per-row commitment term (all f32). One wave per row.
// ---------------------------------------------------------------------------
__global__ __launch_bounds__(256)
void k_project(const float* __restrict__ x, const float* __restrict__ emb,
               const int* __restrict__ idx, float* __restrict__ xq,
               float* __restrict__ scal_out, float* __restrict__ lossterm) {
    const int wave = threadIdx.x >> 5;
    const int lane = threadIdx.x & 31;
    const int row  = blockIdx.x * 8 + wave;
    const int code = idx[row];
    const float* xr = x   + (size_t)row  * DIM + lane * 8;
    const float* cr = emb + (size_t)code * DIM + lane * 8;
    float4 x0 = *(const float4*)(xr);
    float4 x1 = *(const float4*)(xr + 4);
    float4 c0 = *(const float4*)(cr);
    float4 c1 = *(const float4*)(cr + 4);

    float dot = x0.x*c0.x + x0.y*c0.y + x0.z*c0.z + x0.w*c0.w
              + x1.x*c1.x + x1.y*c1.y + x1.z*c1.z + x1.w*c1.w;
    float nsq = c0.x*c0.x + c0.y*c0.y + c0.z*c0.z + c0.w*c0.w
              + c1.x*c1.x + c1.y*c1.y + c1.z*c1.z + c1.w*c1.w;
    float xx  = x0.x*x0.x + x0.y*x0.y + x0.z*x0.z + x0.w*x0.w
              + x1.x*x1.x + x1.y*x1.y + x1.z*x1.z + x1.w*x1.w;
    #pragma unroll
    for (int off = 16; off; off >>= 1) {
        dot += __shfl_xor(dot, off);
        nsq += __shfl_xor(nsq, off);
        xx  += __shfl_xor(xx,  off);
    }
    const float scalar = dot / (nsq + 1e-8f);

    float4 o0, o1;
    o0.x = scalar * c0.x; o0.y = scalar * c0.y; o0.z = scalar * c0.z; o0.w = scalar * c0.w;
    o1.x = scalar * c1.x; o1.y = scalar * c1.y; o1.z = scalar * c1.z; o1.w = scalar * c1.w;
    float* dst = xq + (size_t)row * DIM + lane * 8;
    *(float4*)(dst)     = o0;
    *(float4*)(dst + 4) = o1;

    if (lane == 0) {
        scal_out[row] = scalar;
        const float na = fmaxf(fabsf(scalar) * sqrtf(nsq), 1e-8f);
        const float nb = fmaxf(sqrtf(xx), 1e-8f);
        const float commit = (scalar * dot) / (na * nb);
        lossterm[row] = 1.0f - commit;
    }
}

// ---------------------------------------------------------------------------
// Kernel 5: deterministic single-block reduction of the loss terms.
// ---------------------------------------------------------------------------
__global__ __launch_bounds__(256)
void k_reduce_loss(const float* __restrict__ lossterm, float* __restrict__ out) {
    __shared__ float sm[8];
    const int t = threadIdx.x;
    float s = 0.0f;
    for (int i = t; i < B_ROWS; i += 256) s += lossterm[i];
    #pragma unroll
    for (int off = 16; off; off >>= 1) s += __shfl_xor(s, off);
    const int wave = t >> 5, lane = t & 31;
    if (lane == 0) sm[wave] = s;
    __syncthreads();
    if (t < 8) {
        s = sm[t];
        #pragma unroll
        for (int off = 4; off; off >>= 1) s += __shfl_xor(s, off);
        if (t == 0) out[0] = BETA_C * s / (float)B_ROWS;
    }
}

// ---------------------------------------------------------------------------
extern "C" void kernel_launch(void* const* d_in, const int* in_sizes, int n_in,
                              void* d_out, int out_size, void* d_ws, size_t ws_size,
                              hipStream_t stream) {
    const float* x   = (const float*)d_in[0];   // (B, D)
    const float* emb = (const float*)d_in[1];   // (N, D)

    float* out   = (float*)d_out;
    float* xq    = out;                              // B*D
    float* loss  = out + (size_t)B_ROWS * DIM;       // 1
    float* idx_f = loss + 1;                         // B
    float* scal  = idx_f + B_ROWS;                   // B

    char* ws = (char*)d_ws;
    size_t off = 0;
    unsigned short* cbn = (unsigned short*)(ws + off); off += (size_t)N_CODES * DIM * 2; // 4 MB
    float* partv = (float*)(ws + off); off += (size_t)NSPLIT * B_ROWS * 4;               // 512 KB
    int*   partn = (int*)  (ws + off); off += (size_t)NSPLIT * B_ROWS * 4;               // 512 KB
    int*   idx_i = (int*)  (ws + off); off += (size_t)B_ROWS * 4;                        // 128 KB
    float* lterm = (float*)(ws + off);                                                   // 128 KB

    k_norm_codebook<<<N_CODES / 8, 256, 0, stream>>>(emb, cbn);
    k_sim_argmax  <<<dim3(B_ROWS / 256, NSPLIT), 256, 0, stream>>>(x, cbn, partv, partn);
    k_merge       <<<B_ROWS / 256, 256, 0, stream>>>(partv, partn, idx_i, idx_f);
    k_project     <<<B_ROWS / 8,   256, 0, stream>>>(x, emb, idx_i, xq, scal, lterm);
    k_reduce_loss <<<1,            256, 0, stream>>>(lterm, loss);
}
